// CustomRNN_41566693491393
// MI455X (gfx1250) — compile-verified, matrix operand order checked by reading, not run on hardware
//
#include <hip/hip_runtime.h>
#include <hip/hip_bf16.h>

// ---------------------------------------------------------------------------
// 2-layer LSTM (B=32, T=256, N=M=1024) for MI455X / gfx1250.
//   * bf16 weights/activations (RNE), fp32 accumulate via
//     v_wmma_f32_16x16x32_bf16. All weights (32 MiB bf16) stay L2-resident.
//   * Per step/layer: one WMMA GEMM kernel -> (32 x 4096) gate preacts,
//     then one elementwise LSTM-cell kernel. Stream ordering = recurrence.
//   * GEMM: each wave owns a 32x32 output tile (2 batch x 2 column 16-tiles),
//     weight fragments from L2 (global_load_b128), activation fragments from
//     LDS-staged K-panels (ds_load_b128). 1 fragment load per WMMA.
// ---------------------------------------------------------------------------

typedef __attribute__((ext_vector_type(16))) __bf16 v16bf;
typedef __attribute__((ext_vector_type(8)))  float  v8f;

#define LL 2
#define BB 32
#define TT 256
#define NN 1024
#define MM 1024
#define GG (4 * MM)   // 4096 gate columns
#define KP 256        // K-panel staged in LDS

// ---- helpers --------------------------------------------------------------

__device__ __forceinline__ unsigned short f32_to_bf16_rne(float x) {
    unsigned int bits = __float_as_uint(x);
    unsigned int r = (bits + 0x7FFFu + ((bits >> 16) & 1u)) >> 16;
    return (unsigned short)r;
}

union Frag16 { uint4 u[2]; v16bf v; };
union Acc8   { v8f v; float f[8]; };

// 16x32 bf16 WMMA operand fragment from global memory. `p` points at this
// lane's first element (row*lda + k0 + hi*8): two 16B chunks 32B apart.
__device__ __forceinline__ v16bf load_frag_g(const unsigned short* p) {
    Frag16 f;
    f.u[0] = *(const uint4*)(p);
    f.u[1] = *(const uint4*)(p + 16);
    return f.v;
}

// Same fragment pattern out of an LDS panel (row stride KP elements).
__device__ __forceinline__ v16bf load_frag_s(const unsigned short* p) {
    Frag16 f;
    f.u[0] = *(const uint4*)(p);
    f.u[1] = *(const uint4*)(p + 16);
    return f.v;
}

__device__ __forceinline__ v8f wmma_bf16(v16bf a, v16bf b, v8f c) {
    return __builtin_amdgcn_wmma_f32_16x16x32_bf16(
        false, a, false, b, (short)0, c, false, false);
}

// ---- one-time prep kernels ------------------------------------------------

__global__ void k_f32_to_bf16(const float* __restrict__ src,
                              unsigned short* __restrict__ dst, int n) {
    int i = blockIdx.x * 256 + threadIdx.x;
    if (i < n) dst[i] = f32_to_bf16_rne(src[i]);
}

__global__ void k_bias_sum(const float* __restrict__ a,
                           const float* __restrict__ b,
                           float* __restrict__ o, int n) {
    int i = blockIdx.x * 256 + threadIdx.x;
    if (i < n) o[i] = a[i] + b[i];
}

__global__ void k_init_state(const float* __restrict__ h0,
                             const float* __restrict__ c0,
                             unsigned short* __restrict__ h_bf,
                             float* __restrict__ c_st, int n) {
    int i = blockIdx.x * 256 + threadIdx.x;
    if (i < n) {
        h_bf[i] = f32_to_bf16_rne(h0[i]);
        c_st[i] = c0[i];
    }
}

// ---- gates GEMM -----------------------------------------------------------
// gates(32x4096) = Ain(32xK)*Wih^T + Hprev(32xK)*Whh^T   (K = 1024 each)
// grid = 16 blocks x 256 threads (8 waves). Wave w owns 2 column tiles
// (jt0, jt0+1) x 2 batch tiles. Activations staged per-block in LDS panels.
// ---------------------------------------------------------------------------

__global__ void __launch_bounds__(256)
k_lstm_gates(const unsigned short* __restrict__ Ain,  int lda,  // bf16 input
             const unsigned short* __restrict__ Wih,            // (4096 x 1024)
             const unsigned short* __restrict__ Hprev,          // (32 x 1024)
             const unsigned short* __restrict__ Whh,            // (4096 x 1024)
             float* __restrict__ gates)                         // (32 x 4096)
{
    __shared__ unsigned short sA[BB * KP];   // 16 KB
    __shared__ unsigned short sH[BB * KP];   // 16 KB

    const int tid  = threadIdx.x;
    const int lane = tid & 31;
    const int wave = tid >> 5;
    const int jt0  = (blockIdx.x * 8 + wave) * 2;   // first of 2 column tiles
    const int r    = lane & 15;
    const int hi   = (lane >> 4) & 1;
    const int koff = hi * 8;

    Acc8 acc00, acc01, acc10, acc11;        // [btile][jtile]
    acc00.v = (v8f){}; acc01.v = (v8f){};
    acc10.v = (v8f){}; acc11.v = (v8f){};

    // Per-lane weight row pointers (row = jtile*16 + r of the weight matrix).
    const unsigned short* w0_ih = Wih + (size_t)(jt0 * 16 + r) * NN + koff;
    const unsigned short* w1_ih = w0_ih + (size_t)16 * NN;
    const unsigned short* w0_hh = Whh + (size_t)(jt0 * 16 + r) * MM + koff;
    const unsigned short* w1_hh = w0_hh + (size_t)16 * MM;

    for (int panel = 0; panel < NN / KP; ++panel) {
        const int kp = panel * KP;

        // Cooperative stage: 32 rows x 256 k of Ain and Hprev -> LDS.
        // 1024 uint4 per matrix, 256 threads -> 4 uint4 each.
#pragma unroll
        for (int i = tid; i < (BB * KP) / 8; i += 256) {
            const int row = i >> 5;        // KP/8 = 32 uint4 per row-panel
            const int c8  = (i & 31) * 8;  // element offset within panel row
            *(uint4*)&sA[row * KP + c8] =
                *(const uint4*)(Ain + (size_t)row * lda + kp + c8);
            *(uint4*)&sH[row * KP + c8] =
                *(const uint4*)(Hprev + (size_t)row * MM + kp + c8);
        }
        __syncthreads();

#pragma unroll 4
        for (int kk = 0; kk < KP; kk += 32) {
            const int k0 = kp + kk;

            // input-to-hidden
            {
                v16bf b0 = load_frag_g(w0_ih + k0);
                v16bf b1 = load_frag_g(w1_ih + k0);
                v16bf a0 = load_frag_s(&sA[r        * KP + kk + koff]);
                v16bf a1 = load_frag_s(&sA[(16 + r) * KP + kk + koff]);
                acc00.v = wmma_bf16(a0, b0, acc00.v);
                acc01.v = wmma_bf16(a0, b1, acc01.v);
                acc10.v = wmma_bf16(a1, b0, acc10.v);
                acc11.v = wmma_bf16(a1, b1, acc11.v);
            }
            // hidden-to-hidden
            {
                v16bf b0 = load_frag_g(w0_hh + k0);
                v16bf b1 = load_frag_g(w1_hh + k0);
                v16bf h0 = load_frag_s(&sH[r        * KP + kk + koff]);
                v16bf h1 = load_frag_s(&sH[(16 + r) * KP + kk + koff]);
                acc00.v = wmma_bf16(h0, b0, acc00.v);
                acc01.v = wmma_bf16(h0, b1, acc01.v);
                acc10.v = wmma_bf16(h1, b0, acc10.v);
                acc11.v = wmma_bf16(h1, b1, acc11.v);
            }
        }
        __syncthreads();
    }

    // C/D layout: VGPR v holds (tile row v + 8*hi, tile col r).
    const int j0 = jt0 * 16 + r;
    const int j1 = j0 + 16;
#pragma unroll
    for (int v = 0; v < 8; ++v) {
        const int b = v + hi * 8;
        gates[(size_t)b        * GG + j0] = acc00.f[v];
        gates[(size_t)b        * GG + j1] = acc01.f[v];
        gates[(size_t)(16 + b) * GG + j0] = acc10.f[v];
        gates[(size_t)(16 + b) * GG + j1] = acc11.f[v];
    }
}

// ---- elementwise LSTM cell ------------------------------------------------

__global__ void __launch_bounds__(256)
k_lstm_cell(const float* __restrict__ gates,   // (32 x 4096)
            const float* __restrict__ bsum,    // (4096) = bih + bhh
            float* __restrict__ c_state,       // (32 x 1024)
            unsigned short* __restrict__ h_bf, // (32 x 1024) bf16 out
            float* __restrict__ outs,          // d_out outs base or nullptr
            int t,
            float* __restrict__ h_fin,
            float* __restrict__ c_fin,
            int last)
{
    const int idx = blockIdx.x * 256 + threadIdx.x;   // 0..32767
    const int b = idx >> 10;
    const int m = idx & (MM - 1);
    const float* g = gates + (size_t)b * GG;

    float iv = g[m]          + bsum[m];
    float fv = g[MM + m]     + bsum[MM + m];
    float gv = g[2 * MM + m] + bsum[2 * MM + m];
    float ov = g[3 * MM + m] + bsum[3 * MM + m];

    float is = 1.0f / (1.0f + __expf(-iv));
    float fs = 1.0f / (1.0f + __expf(-fv));
    float os = 1.0f / (1.0f + __expf(-ov));
    float gt = tanhf(gv);

    float cn = fs * c_state[idx] + is * gt;
    float hn = os * tanhf(cn);

    c_state[idx] = cn;
    h_bf[idx]    = f32_to_bf16_rne(hn);

    if (outs) outs[(size_t)b * (TT * MM) + (size_t)t * MM + m] = hn;
    if (last) { h_fin[idx] = hn; c_fin[idx] = cn; }
}

// ---- host-side orchestration ----------------------------------------------

extern "C" void kernel_launch(void* const* d_in, const int* in_sizes, int n_in,
                              void* d_out, int out_size, void* d_ws, size_t ws_size,
                              hipStream_t stream) {
    const float* x   = (const float*)d_in[0];  // (B, T, N)
    const float* h0  = (const float*)d_in[1];  // (L, B, M)
    const float* c0  = (const float*)d_in[2];  // (L, B, M)
    const float* Wih = (const float*)d_in[3];  // (L, 4M, N)
    const float* Whh = (const float*)d_in[4];  // (L, 4M, M)
    const float* bih = (const float*)d_in[5];  // (L, 4M)
    const float* bhh = (const float*)d_in[6];  // (L, 4M)

    char* ws = (char*)d_ws;
    size_t off = 0;
    auto carve = [&](size_t bytes) {
        void* p = ws + off;
        off = (off + bytes + 255) & ~(size_t)255;
        return p;
    };
    unsigned short* Wih_b = (unsigned short*)carve((size_t)LL * GG * NN * 2);
    unsigned short* Whh_b = (unsigned short*)carve((size_t)LL * GG * MM * 2);
    unsigned short* x_b   = (unsigned short*)carve((size_t)BB * TT * NN * 2);
    unsigned short* h_b   = (unsigned short*)carve((size_t)LL * BB * MM * 2);
    float*          c_st  = (float*)carve((size_t)LL * BB * MM * 4);
    float*          gates = (float*)carve((size_t)BB * GG * 4);
    float*          bsum  = (float*)carve((size_t)LL * GG * 4);

    const int nW = LL * GG * NN;
    k_f32_to_bf16<<<(nW + 255) / 256, 256, 0, stream>>>(Wih, Wih_b, nW);
    k_f32_to_bf16<<<(nW + 255) / 256, 256, 0, stream>>>(Whh, Whh_b, nW);
    const int nX = BB * TT * NN;
    k_f32_to_bf16<<<(nX + 255) / 256, 256, 0, stream>>>(x, x_b, nX);
    const int nH = LL * BB * MM;
    k_init_state<<<(nH + 255) / 256, 256, 0, stream>>>(h0, c0, h_b, c_st, nH);
    const int nBias = LL * GG;
    k_bias_sum<<<(nBias + 255) / 256, 256, 0, stream>>>(bih, bhh, bsum, nBias);

    float* outp = (float*)d_out;                       // (B, T, M)
    float* hfin = outp + (size_t)BB * TT * MM;         // (L, B, M)
    float* cfin = hfin + (size_t)LL * BB * MM;         // (L, B, M)

    for (int t = 0; t < TT; ++t) {
        const int last = (t == TT - 1);

        // Layer 0: input = x[:, t, :] (row stride T*N in bf16 buffer).
        k_lstm_gates<<<16, 256, 0, stream>>>(
            x_b + (size_t)t * NN, TT * NN,
            Wih_b, h_b, Whh_b, gates);
        k_lstm_cell<<<128, 256, 0, stream>>>(
            gates, bsum, c_st, h_b,
            nullptr, t, hfin, cfin, last);

        // Layer 1: input = layer-0 h just produced (row stride M).
        k_lstm_gates<<<16, 256, 0, stream>>>(
            h_b, MM,
            Wih_b + (size_t)GG * NN,
            h_b + (size_t)BB * MM,
            Whh_b + (size_t)GG * MM, gates);
        k_lstm_cell<<<128, 256, 0, stream>>>(
            gates, bsum + GG, c_st + (size_t)BB * MM, h_b + (size_t)BB * MM,
            outp, t, hfin + (size_t)BB * MM, cfin + (size_t)BB * MM, last);
    }
}
